// ATT_NLM_86320252715608
// MI455X (gfx1250) — matrix operand order repaired and checked by reading, not compiled, not run on hardware
//
#include <hip/hip_runtime.h>
#include <math.h>

typedef _Float16 v8h  __attribute__((ext_vector_type(8)));
typedef _Float16 v16h __attribute__((ext_vector_type(16)));
typedef float    v8f  __attribute__((ext_vector_type(8)));
typedef float    v4f  __attribute__((ext_vector_type(4)));

#define B_   16
#define S1_  61
#define S_   3721
#define NW_  61
#define WIN_ 61
#define D_   128
#define H_   8
#define L_   4
#define FF_  512
#define BS_  59536   /* B_*S_ */

static __device__ __forceinline__ v16h cat8(v8h lo, v8h hi) {
  return __builtin_shufflevector(lo, hi, 0,1,2,3,4,5,6,7,8,9,10,11,12,13,14,15);
}

static __device__ __forceinline__ float gelu_f(float x) {
  return 0.5f * x * (1.0f + erff(x * 0.70710678118654752f));
}

// ---------------------------------------------------------------------------
// Weight transpose + f32->f16 convert:  WT[n*K + k] = (f16)W[k*N + n]
// ---------------------------------------------------------------------------
__global__ __launch_bounds__(256)
void tc_kernel(const float* __restrict__ W, _Float16* __restrict__ WT, int K, int N) {
  const int idx = blockIdx.x * 256 + threadIdx.x;
  if (idx >= K * N) return;
  const int k = idx / N;
  const int n = idx - k * N;
  WT[(long)n * K + k] = (_Float16)W[idx];
}

// ---------------------------------------------------------------------------
// Embedding: 7x7 conv (49 ch) + ReLU + linear 49->128 + bias + pos table
// One block (128 threads) per pixel.
// ---------------------------------------------------------------------------
__global__ __launch_bounds__(128)
void embed_kernel(const float* __restrict__ X, const float* __restrict__ CW,
                  const float* __restrict__ CB, const float* __restrict__ LW,
                  const float* __restrict__ LB, const float* __restrict__ POS,
                  float* __restrict__ Hout)
{
  __shared__ float patch[49];
  __shared__ float pch[49];
  const int pix = blockIdx.x;
  const int b = pix / S_;
  const int s = pix - b * S_;
  const int py = s / S1_;
  const int px = s - py * S1_;
  const int t = threadIdx.x;
  if (t < 49) {
    const int dy = t / 7 - 3, dx = t % 7 - 3;
    const int yy = py + dy, xx = px + dx;
    patch[t] = (yy >= 0 && yy < S1_ && xx >= 0 && xx < S1_)
             ? X[((long)b * S1_ + yy) * S1_ + xx] : 0.0f;
  }
  __syncthreads();
  if (t < 49) {
    float a = CB[t];
    const float* wr = CW + t * 49;
    for (int i = 0; i < 49; ++i) a += patch[i] * wr[i];
    pch[t] = fmaxf(a, 0.0f);
  }
  __syncthreads();
  float a = LB[t] + POS[(long)s * D_ + t];
  for (int c = 0; c < 49; ++c) a += pch[c] * LW[c * D_ + t];
  Hout[(long)pix * D_ + t] = a;
}

// ---------------------------------------------------------------------------
// LayerNorm (one wave per row of 128), writes f16
// ---------------------------------------------------------------------------
__global__ __launch_bounds__(256)
void ln_kernel(const float* __restrict__ X, const float* __restrict__ sc,
               const float* __restrict__ bi, _Float16* __restrict__ Y, int M)
{
  const int row = blockIdx.x * 8 + (threadIdx.x >> 5);
  if (row >= M) return;
  const int lane = threadIdx.x & 31;
  const v4f xv = *(const v4f*)(X + (long)row * D_ + lane * 4);
  float s = xv[0] + xv[1] + xv[2] + xv[3];
#pragma unroll
  for (int o = 16; o > 0; o >>= 1) s += __shfl_xor(s, o, 32);
  const float m = s * (1.0f / 128.0f);
  const float d0 = xv[0]-m, d1 = xv[1]-m, d2 = xv[2]-m, d3 = xv[3]-m;
  float vs = d0*d0 + d1*d1 + d2*d2 + d3*d3;
#pragma unroll
  for (int o = 16; o > 0; o >>= 1) vs += __shfl_xor(vs, o, 32);
  const float rstd = rsqrtf(vs * (1.0f / 128.0f) + 1e-5f);
  const int c = lane * 4;
  _Float16* yp = Y + (long)row * D_ + c;
  yp[0] = (_Float16)(d0 * rstd * sc[c+0] + bi[c+0]);
  yp[1] = (_Float16)(d1 * rstd * sc[c+1] + bi[c+1]);
  yp[2] = (_Float16)(d2 * rstd * sc[c+2] + bi[c+2]);
  yp[3] = (_Float16)(d3 * rstd * sc[c+3] + bi[c+3]);
}

// ---------------------------------------------------------------------------
// f32 -> f16 convert (4 elements per thread)
// ---------------------------------------------------------------------------
__global__ __launch_bounds__(256)
void cvt16_kernel(const float* __restrict__ X, _Float16* __restrict__ Y, int n4) {
  const int idx = blockIdx.x * 256 + threadIdx.x;
  if (idx >= n4) return;
  const v4f v = *(const v4f*)(X + (long)idx * 4);
  _Float16* yp = Y + (long)idx * 4;
  yp[0] = (_Float16)v[0]; yp[1] = (_Float16)v[1];
  yp[2] = (_Float16)v[2]; yp[3] = (_Float16)v[3];
}

// ---------------------------------------------------------------------------
// Generic WMMA GEMM:  C[M,128] = A16[M,128] @ BT16[128,128]^T  (BT is [N,K])
// One wave computes 16 rows x 128 cols (32 v_wmma_f32_16x16x32_f16).
// A-tile 16x32 lane layout: lane l -> row l%16, K = (l/16)*8 + {0..7, 16..23}
// B-tile 32x16 lane layout: lane l -> col l%16, K = (l/16)*8 + {0..7, 16..23}
// C-tile: lane l -> col l%16, rows 8*(l/16)+r (r = vgpr 0..7)
// ---------------------------------------------------------------------------
template<bool BIAS, bool RESID, int ACT, bool OUT16, bool OUT32>
__global__ __launch_bounds__(256)
void gemm128_kernel(const _Float16* __restrict__ A, const _Float16* __restrict__ BT,
                    const float* __restrict__ bias, float* __restrict__ C32,
                    _Float16* __restrict__ C16, int M)
{
  const int wave = threadIdx.x >> 5;
  const int lane = threadIdx.x & 31;
  const int m0 = (blockIdx.x * 8 + wave) * 16;
  if (m0 >= M) return;
  const int nlo  = lane & 15;
  const int hi16 = lane >> 4;
  const int kA = hi16 * 8;
  const int kB = hi16 * 8;

  v16h a[4];
  {
    const _Float16* ar = A + (long)(m0 + nlo) * D_;
#pragma unroll
    for (int kt = 0; kt < 4; ++kt)
      a[kt] = cat8(*(const v8h*)(ar + kt*32 + kA),
                   *(const v8h*)(ar + kt*32 + kA + 16));
  }
  const v8f vz = {0,0,0,0,0,0,0,0};
  v8f acc[8] = {vz,vz,vz,vz,vz,vz,vz,vz};
#pragma unroll
  for (int nt = 0; nt < 8; ++nt) {
    const _Float16* br = BT + (long)(nt*16 + nlo) * D_;
#pragma unroll
    for (int kt = 0; kt < 4; ++kt) {
      v16h bm = cat8(*(const v8h*)(br + kt*32 + kB),
                     *(const v8h*)(br + kt*32 + kB + 16));
      acc[nt] = __builtin_amdgcn_wmma_f32_16x16x32_f16(false, a[kt], false, bm,
                                                       (short)0, acc[nt], false, false);
    }
  }
  const int rb = hi16 * 8;
#pragma unroll
  for (int nt = 0; nt < 8; ++nt) {
    const int col = nt*16 + nlo;
    float bv = 0.0f;
    if constexpr (BIAS) bv = bias[col];
#pragma unroll
    for (int r = 0; r < 8; ++r) {
      const long idx = (long)(m0 + rb + r) * D_ + col;
      float v = acc[nt][r] + bv;
      if constexpr (RESID) v += C32[idx];
      if constexpr (ACT == 1) v = fmaxf(v, 0.0f);
      if constexpr (OUT32) C32[idx] = v;
      if constexpr (OUT16) C16[idx] = (_Float16)v;
    }
  }
}

// ---------------------------------------------------------------------------
// Fused FFN:  H += gelu(A16 @ W1 + b1) @ W2 + b2   (W1T:[512,128], W2T:[128,512])
// One wave keeps 16 rows resident; loops 16 chunks of 32 FF dims; C->A layout
// conversion for the FF2 operand via per-wave LDS (ordered with s_wait_dscnt).
// ---------------------------------------------------------------------------
__global__ __launch_bounds__(256)
void ffn_kernel(const _Float16* __restrict__ A, const _Float16* __restrict__ W1T,
                const float* __restrict__ b1, const _Float16* __restrict__ W2T,
                const float* __restrict__ b2, float* __restrict__ H, int M)
{
  __shared__ _Float16 tls[8][16 * 32];
  const int wave = threadIdx.x >> 5;
  const int lane = threadIdx.x & 31;
  const int m0 = (blockIdx.x * 8 + wave) * 16;
  if (m0 >= M) return;
  const int nlo  = lane & 15;
  const int hi16 = lane >> 4;
  const int kA = hi16 * 8;
  const int kB = hi16 * 8;
  _Float16* tb = tls[wave];

  v16h a[4];
  {
    const _Float16* ar = A + (long)(m0 + nlo) * D_;
#pragma unroll
    for (int kt = 0; kt < 4; ++kt)
      a[kt] = cat8(*(const v8h*)(ar + kt*32 + kA),
                   *(const v8h*)(ar + kt*32 + kA + 16));
  }
  const v8f vz = {0,0,0,0,0,0,0,0};
  v8f acc[8] = {vz,vz,vz,vz,vz,vz,vz,vz};
  const int rb = hi16 * 8;

  for (int c = 0; c < 16; ++c) {
    v8f t0 = vz, t1 = vz;
    const _Float16* b0r = W1T + (long)(c*32 + nlo) * D_;
    const _Float16* b1r = W1T + (long)(c*32 + 16 + nlo) * D_;
#pragma unroll
    for (int kt = 0; kt < 4; ++kt) {
      v16h bm0 = cat8(*(const v8h*)(b0r + kt*32 + kB), *(const v8h*)(b0r + kt*32 + kB + 16));
      t0 = __builtin_amdgcn_wmma_f32_16x16x32_f16(false, a[kt], false, bm0, (short)0, t0, false, false);
      v16h bm1 = cat8(*(const v8h*)(b1r + kt*32 + kB), *(const v8h*)(b1r + kt*32 + kB + 16));
      t1 = __builtin_amdgcn_wmma_f32_16x16x32_f16(false, a[kt], false, bm1, (short)0, t1, false, false);
    }
    const float ba = b1[c*32 + nlo];
    const float bb = b1[c*32 + 16 + nlo];
#pragma unroll
    for (int r = 0; r < 8; ++r) {
      const int rr = rb + r;
      tb[rr*32 + nlo]      = (_Float16)gelu_f(t0[r] + ba);
      tb[rr*32 + 16 + nlo] = (_Float16)gelu_f(t1[r] + bb);
    }
    asm volatile("s_wait_dscnt 0" ::: "memory");   // wave-private LDS: order store->load
    v16h ta = cat8(*(const v8h*)(tb + nlo*32 + kA),
                   *(const v8h*)(tb + nlo*32 + kA + 16));
#pragma unroll
    for (int nt = 0; nt < 8; ++nt) {
      const _Float16* wr = W2T + (long)(nt*16 + nlo) * FF_ + c*32;
      v16h bm = cat8(*(const v8h*)(wr + kB), *(const v8h*)(wr + kB + 16));
      acc[nt] = __builtin_amdgcn_wmma_f32_16x16x32_f16(false, ta, false, bm, (short)0, acc[nt], false, false);
    }
    asm volatile("" ::: "memory");                 // keep loads before next chunk's stores
  }
#pragma unroll
  for (int nt = 0; nt < 8; ++nt) {
    const int col = nt*16 + nlo;
    const float bv = b2[col];
#pragma unroll
    for (int r = 0; r < 8; ++r) {
      const long idx = (long)(m0 + rb + r) * D_ + col;
      H[idx] += acc[nt][r] + bv;
    }
  }
}

// ---------------------------------------------------------------------------
// Local windowed attention (look back/fwd 1, window 61), online softmax.
// Block = (b, window); wave = head; lane owns query rows lane and lane+32.
// ---------------------------------------------------------------------------
__global__ __launch_bounds__(256)
void attn_kernel(const _Float16* __restrict__ Q, const _Float16* __restrict__ Km,
                 const _Float16* __restrict__ Vm, _Float16* __restrict__ O)
{
  const int blk = blockIdx.x;
  const int b = blk / NW_;
  const int w = blk - b * NW_;
  const int hd = threadIdx.x >> 5;
  const int lane = threadIdx.x & 31;
  const long base = (long)b * S_ * D_ + hd * 16;
  const int r0 = lane;
  const int r1 = lane + 32;
  const bool has1 = (r1 < WIN_);

  float q0[16], q1[16], o0[16], o1[16];
  {
    const v8h* qp = (const v8h*)(Q + base + (long)(w * WIN_ + r0) * D_);
    const v8h lo = qp[0], hi = qp[1];
#pragma unroll
    for (int i = 0; i < 8; ++i) { q0[i] = (float)lo[i] * 0.25f; q0[8+i] = (float)hi[i] * 0.25f; }
  }
  if (has1) {
    const v8h* qp = (const v8h*)(Q + base + (long)(w * WIN_ + r1) * D_);
    const v8h lo = qp[0], hi = qp[1];
#pragma unroll
    for (int i = 0; i < 8; ++i) { q1[i] = (float)lo[i] * 0.25f; q1[8+i] = (float)hi[i] * 0.25f; }
  } else {
#pragma unroll
    for (int i = 0; i < 16; ++i) q1[i] = 0.0f;
  }
#pragma unroll
  for (int i = 0; i < 16; ++i) { o0[i] = 0.0f; o1[i] = 0.0f; }
  float mA = -1e30f, mB = -1e30f, lA = 0.0f, lB = 0.0f;

  for (int j = 0; j < 3 * WIN_; ++j) {
    const int kp = (w - 1) * WIN_ + j;       // uniform across wave
    if (kp < 0 || kp >= S_) continue;        // masked (padded) windows
    const v8h* kr = (const v8h*)(Km + base + (long)kp * D_);
    const v8h klo = kr[0], khi = kr[1];
    const v8h* vr = (const v8h*)(Vm + base + (long)kp * D_);
    const v8h vlo = vr[0], vhi = vr[1];
    float s0 = 0.0f, s1 = 0.0f;
#pragma unroll
    for (int i = 0; i < 8; ++i) {
      const float kv = (float)klo[i];
      s0 += q0[i] * kv; s1 += q1[i] * kv;
    }
#pragma unroll
    for (int i = 0; i < 8; ++i) {
      const float kv = (float)khi[i];
      s0 += q0[8+i] * kv; s1 += q1[8+i] * kv;
    }
    const float nmA = fmaxf(mA, s0);
    const float cA = __expf(mA - nmA);
    const float pA = __expf(s0 - nmA);
    lA = lA * cA + pA; mA = nmA;
    const float nmB = fmaxf(mB, s1);
    const float cB = __expf(mB - nmB);
    const float pB = __expf(s1 - nmB);
    lB = lB * cB + pB; mB = nmB;
#pragma unroll
    for (int i = 0; i < 8; ++i) {
      const float va = (float)vlo[i], vb = (float)vhi[i];
      o0[i]   = o0[i]   * cA + pA * va;
      o0[8+i] = o0[8+i] * cA + pA * vb;
      o1[i]   = o1[i]   * cB + pB * va;
      o1[8+i] = o1[8+i] * cB + pB * vb;
    }
  }
  {
    const float inv = 1.0f / lA;
    v8h lo, hi;
#pragma unroll
    for (int i = 0; i < 8; ++i) { lo[i] = (_Float16)(o0[i] * inv); hi[i] = (_Float16)(o0[8+i] * inv); }
    v8h* op = (v8h*)(O + base + (long)(w * WIN_ + r0) * D_);
    op[0] = lo; op[1] = hi;
  }
  if (has1) {
    const float inv = 1.0f / lB;
    v8h lo, hi;
#pragma unroll
    for (int i = 0; i < 8; ++i) { lo[i] = (_Float16)(o1[i] * inv); hi[i] = (_Float16)(o1[8+i] * inv); }
    v8h* op = (v8h*)(O + base + (long)(w * WIN_ + r1) * D_);
    op[0] = lo; op[1] = hi;
  }
}

// ---------------------------------------------------------------------------
// Final head: out[row] = T16[row,:] . pre_w2 + pre_b2   (one wave per row)
// ---------------------------------------------------------------------------
__global__ __launch_bounds__(256)
void head_kernel(const _Float16* __restrict__ T, const float* __restrict__ w2,
                 const float* __restrict__ b2, float* __restrict__ out, int M)
{
  const int row = blockIdx.x * 8 + (threadIdx.x >> 5);
  if (row >= M) return;
  const int lane = threadIdx.x & 31;
  const _Float16* tp = T + (long)row * D_ + lane * 4;
  const float* wp = w2 + lane * 4;
  float s = (float)tp[0]*wp[0] + (float)tp[1]*wp[1] + (float)tp[2]*wp[2] + (float)tp[3]*wp[3];
#pragma unroll
  for (int o = 16; o > 0; o >>= 1) s += __shfl_xor(s, o, 32);
  if (lane == 0) out[row] = s + b2[0];
}

// ---------------------------------------------------------------------------
extern "C" void kernel_launch(void* const* d_in, const int* in_sizes, int n_in,
                              void* d_out, int out_size, void* d_ws, size_t ws_size,
                              hipStream_t stream) {
  (void)in_sizes; (void)n_in; (void)out_size; (void)ws_size;
  const float* x      = (const float*)d_in[0];
  const float* conv_w = (const float*)d_in[1];
  const float* conv_b = (const float*)d_in[2];
  const float* lt_w   = (const float*)d_in[3];
  const float* lt_b   = (const float*)d_in[4];
  const float* pos    = (const float*)d_in[5];
  const float* ln1_s  = (const float*)d_in[6];
  const float* ln1_b  = (const float*)d_in[7];
  const float* wq     = (const float*)d_in[8];
  const float* wk     = (const float*)d_in[9];
  const float* wv     = (const float*)d_in[10];
  const float* wo     = (const float*)d_in[11];
  const float* wo_b   = (const float*)d_in[12];
  const float* ln2_s  = (const float*)d_in[13];
  const float* ln2_b  = (const float*)d_in[14];
  const float* ffw1   = (const float*)d_in[15];
  const float* ffb1   = (const float*)d_in[16];
  const float* ffw2   = (const float*)d_in[17];
  const float* ffb2   = (const float*)d_in[18];
  const float* pw1    = (const float*)d_in[19];
  const float* pb1    = (const float*)d_in[20];
  const float* pw2    = (const float*)d_in[21];
  const float* pb2    = (const float*)d_in[22];

  // workspace layout
  const size_t F32ROW = (size_t)BS_ * D_ * sizeof(float);     // 30.5 MB
  const size_t F16ROW = (size_t)BS_ * D_ * sizeof(_Float16);  // 15.2 MB
  char* wsb = (char*)d_ws;
  float*    h   = (float*)(wsb);
  _Float16* a16 = (_Float16*)(wsb + F32ROW);
  _Float16* q16 = (_Float16*)(wsb + F32ROW + 1*F16ROW);
  _Float16* k16 = (_Float16*)(wsb + F32ROW + 2*F16ROW);
  _Float16* v16 = (_Float16*)(wsb + F32ROW + 3*F16ROW);
  _Float16* o16 = (_Float16*)(wsb + F32ROW + 4*F16ROW);
  _Float16* wp  = (_Float16*)(wsb + F32ROW + 5*F16ROW);       // packed f16 weights

  const int WL = 196608;  // halves per layer: 4*16384 + 65536 + 65536

  // 1) convert + transpose all weights to f16 [N,K]
  for (int l = 0; l < L_; ++l) {
    _Float16* base = wp + (size_t)l * WL;
    tc_kernel<<<64,  256, 0, stream>>>(wq   + l*16384, base,           128, 128);
    tc_kernel<<<64,  256, 0, stream>>>(wk   + l*16384, base + 16384,   128, 128);
    tc_kernel<<<64,  256, 0, stream>>>(wv   + l*16384, base + 32768,   128, 128);
    tc_kernel<<<64,  256, 0, stream>>>(wo   + l*16384, base + 49152,   128, 128);
    tc_kernel<<<256, 256, 0, stream>>>(ffw1 + l*65536, base + 65536,   128, 512);
    tc_kernel<<<256, 256, 0, stream>>>(ffw2 + l*65536, base + 131072,  512, 128);
  }
  tc_kernel<<<64, 256, 0, stream>>>(pw1, wp + (size_t)4*WL, 128, 128);

  // 2) embedding
  embed_kernel<<<BS_, 128, 0, stream>>>(x, conv_w, conv_b, lt_w, lt_b, pos, h);

  const int GB = (BS_/16 + 7) / 8;  // 466 blocks for 16-row waves
  const int RB = (BS_ + 7) / 8;     // 7442 blocks for 1-row waves

  // 3) transformer layers
  for (int l = 0; l < L_; ++l) {
    const _Float16* base = wp + (size_t)l * WL;
    ln_kernel<<<RB, 256, 0, stream>>>(h, ln1_s + l*D_, ln1_b + l*D_, a16, BS_);
    gemm128_kernel<false,false,0,true,false><<<GB,256,0,stream>>>(a16, base,         nullptr, nullptr, q16, BS_);
    gemm128_kernel<false,false,0,true,false><<<GB,256,0,stream>>>(a16, base + 16384, nullptr, nullptr, k16, BS_);
    gemm128_kernel<false,false,0,true,false><<<GB,256,0,stream>>>(a16, base + 32768, nullptr, nullptr, v16, BS_);
    attn_kernel<<<B_*NW_, 256, 0, stream>>>(q16, k16, v16, o16);
    gemm128_kernel<true,true,0,false,true><<<GB,256,0,stream>>>(o16, base + 49152, wo_b + l*D_, h, nullptr, BS_);
    ln_kernel<<<RB, 256, 0, stream>>>(h, ln2_s + l*D_, ln2_b + l*D_, a16, BS_);
    ffn_kernel<<<GB, 256, 0, stream>>>(a16, base + 65536, ffb1 + l*FF_, base + 131072, ffb2 + l*D_, h, BS_);
  }

  // 4) final MLP head
  cvt16_kernel<<<RB, 256, 0, stream>>>(h, a16, BS_ * D_ / 4);
  gemm128_kernel<true,false,1,true,false><<<GB,256,0,stream>>>(a16, wp + (size_t)4*WL, pb1, nullptr, q16, BS_);
  head_kernel<<<RB, 256, 0, stream>>>(q16, pw2, pb2, (float*)d_out, BS_);
}
